// MaskedAttention_18210661335200
// MI455X (gfx1250) — matrix operand order repaired and checked by reading, not compiled, not run on hardware
//
#include <hip/hip_runtime.h>
#include <hip/hip_bf16.h>
#include <math.h>

// Problem constants (B,T,C,H = 4,2048,1024,16; D=64)
#define BB 4
#define TT 2048
#define CC 1024
#define HH 16
#define DD 64
#define N3C (3 * CC)   // 3072
#define MTOT (BB * TT) // 8192

typedef __attribute__((ext_vector_type(16))) _Float16 v16h;
typedef __attribute__((ext_vector_type(8)))  _Float16 v8h;
typedef __attribute__((ext_vector_type(8)))  float    v8f;

static __device__ __forceinline__ v16h cat16(v8h lo, v8h hi) {
    v16h r;
#pragma unroll
    for (int i = 0; i < 8; ++i) { r[i] = lo[i]; r[i + 8] = hi[i]; }
    return r;
}

// ---------------------------------------------------------------------------
// Kernel 1: fp32 -> f16 conversion of x (8,388,608 elements, 4 per thread)
// ---------------------------------------------------------------------------
__global__ void cvt_x_kernel(const float* __restrict__ x, _Float16* __restrict__ xh) {
    int i = (blockIdx.x * blockDim.x + threadIdx.x) * 4;
    float4 f = *(const float4*)(x + i);
    xh[i + 0] = (_Float16)f.x;
    xh[i + 1] = (_Float16)f.y;
    xh[i + 2] = (_Float16)f.z;
    xh[i + 3] = (_Float16)f.w;
}

// ---------------------------------------------------------------------------
// Kernel 2: transpose + convert W [C x 3C] fp32  ->  Wt [3C x C] f16
// (so GEMM B-fragments are K-contiguous per lane)
// ---------------------------------------------------------------------------
__global__ void cvt_wt_kernel(const float* __restrict__ w, _Float16* __restrict__ wt) {
    int idx = blockIdx.x * blockDim.x + threadIdx.x; // over 3C*C
    int n  = idx >> 10;        // output row  (0..3071)
    int kk = idx & (CC - 1);   // output col  (0..1023)
    wt[idx] = (_Float16)w[kk * N3C + n];
}

// ---------------------------------------------------------------------------
// Kernel 3: QKV projection GEMM with WMMA. K-loop unrolled x2 with ping-pong
//   fragment register sets (no rotation copies): phase A's 8 WMMAs overlap
//   phase B's loads and vice versa.
//   qkv = xh[8192x1024] @ Wt^T + bias; columns split [0,1024)=K, [1024,2048)=Q,
//   [2048,3072)=V.  K,Q stored [B,H,T,D] f16; V stored transposed [B,H,D,T].
//   Block = 128 threads (4 waves); wave tile = 32(M) x 64(N); grid = (64,48).
// ---------------------------------------------------------------------------
__global__ void
__launch_bounds__(128)
qkv_gemm_kernel(const _Float16* __restrict__ xh, const _Float16* __restrict__ wt,
                const float* __restrict__ bias,
                _Float16* __restrict__ kbuf, _Float16* __restrict__ qbuf,
                _Float16* __restrict__ vtbuf) {
    const int lane   = threadIdx.x & 31;
    const int wave   = threadIdx.x >> 5;
    const int lane16 = lane & 15;
    const int offA   = (lane < 16) ? 0 : 16;  // A-fragment half-offset
    const int offB   = (lane < 16) ? 0 : 16;  // B-fragment K-offset
    const int m0 = (blockIdx.x * 4 + wave) * 32;
    const int n0 = blockIdx.y * 64;

    v8f acc[2][4];
#pragma unroll
    for (int i = 0; i < 2; ++i)
#pragma unroll
        for (int j = 0; j < 4; ++j)
#pragma unroll
            for (int e = 0; e < 8; ++e) acc[i][j][e] = 0.0f;

    const _Float16* arow0 = xh + (size_t)(m0 + lane16) * CC;
    const _Float16* arow1 = arow0 + (size_t)16 * CC;
    const _Float16* brow[4];
#pragma unroll
    for (int j = 0; j < 4; ++j)
        brow[j] = wt + (size_t)(n0 + j * 16 + lane16) * CC;

    // Ping-pong fragment sets
    v16h aA0, aA1, bfA[4];   // even phase (kb)
    v16h aB0, aB1, bfB[4];   // odd phase (kb+32)

    // Prologue: even-phase fragments for kb = 0
    aA0 = cat16(*(const v8h*)(arow0 + offA), *(const v8h*)(arow0 + 8 + offA));
    aA1 = cat16(*(const v8h*)(arow1 + offA), *(const v8h*)(arow1 + 8 + offA));
#pragma unroll
    for (int j = 0; j < 4; ++j) bfA[j] = *(const v16h*)(brow[j] + offB);

    for (int kb = 0; kb < CC; kb += 64) {
        const int k1 = kb + 32;   // always < CC (CC % 64 == 0)
        const int k2 = kb + 64;

        // Load odd-phase fragments (k1) -- overlaps even-phase WMMAs below.
        __builtin_prefetch(arow0 + k1 + 64, 0, 3);
        __builtin_prefetch(brow[0] + k1 + 64, 0, 3);
        aB0 = cat16(*(const v8h*)(arow0 + k1 + offA),
                    *(const v8h*)(arow0 + k1 + 8 + offA));
        aB1 = cat16(*(const v8h*)(arow1 + k1 + offA),
                    *(const v8h*)(arow1 + k1 + 8 + offA));
#pragma unroll
        for (int j = 0; j < 4; ++j)
            bfB[j] = *(const v16h*)(brow[j] + k1 + offB);

        // Even-phase WMMAs (kb)
#pragma unroll
        for (int j = 0; j < 4; ++j) {
            acc[0][j] = __builtin_amdgcn_wmma_f32_16x16x32_f16(
                false, aA0, false, bfA[j], (short)0, acc[0][j], false, false);
            acc[1][j] = __builtin_amdgcn_wmma_f32_16x16x32_f16(
                false, aA1, false, bfA[j], (short)0, acc[1][j], false, false);
        }

        // Load next even-phase fragments (k2) -- overlaps odd-phase WMMAs.
        if (k2 < CC) {
            aA0 = cat16(*(const v8h*)(arow0 + k2 + offA),
                        *(const v8h*)(arow0 + k2 + 8 + offA));
            aA1 = cat16(*(const v8h*)(arow1 + k2 + offA),
                        *(const v8h*)(arow1 + k2 + 8 + offA));
#pragma unroll
            for (int j = 0; j < 4; ++j)
                bfA[j] = *(const v16h*)(brow[j] + k2 + offB);
        }

        // Odd-phase WMMAs (k1)
#pragma unroll
        for (int j = 0; j < 4; ++j) {
            acc[0][j] = __builtin_amdgcn_wmma_f32_16x16x32_f16(
                false, aB0, false, bfB[j], (short)0, acc[0][j], false, false);
            acc[1][j] = __builtin_amdgcn_wmma_f32_16x16x32_f16(
                false, aB1, false, bfB[j], (short)0, acc[1][j], false, false);
        }
    }

    // Bias (column n0+j*16+lane16, same for all 8 rows of a lane)
#pragma unroll
    for (int j = 0; j < 4; ++j) {
        float bj = bias[n0 + j * 16 + lane16];
#pragma unroll
        for (int i = 0; i < 2; ++i)
#pragma unroll
            for (int e = 0; e < 8; ++e) acc[i][j][e] += bj;
    }

    // Route output: n0 is a multiple of 64 => whole 64-col tile is one (sec,h)
    const int sec = n0 >> 10;          // 0=k, 1=q, 2=v (reference split order)
    const int h   = (n0 & 1023) >> 6;  // head
    const int rhi = (lane < 16) ? 0 : 8;
#pragma unroll
    for (int i = 0; i < 2; ++i) {
#pragma unroll
        for (int e = 0; e < 8; ++e) {
            int m = m0 + i * 16 + e + rhi;  // global token row
            int b = m >> 11;                // / 2048
            int t = m & (TT - 1);
            size_t bh = (size_t)(b * HH + h);
#pragma unroll
            for (int j = 0; j < 4; ++j) {
                int d = j * 16 + lane16;
                _Float16 hv = (_Float16)acc[i][j][e];
                if (sec == 0)      kbuf[(bh * TT + t) * DD + d] = hv;
                else if (sec == 1) qbuf[(bh * TT + t) * DD + d] = hv;
                else               vtbuf[(bh * DD + d) * TT + t] = hv;
            }
        }
    }
}

// ---------------------------------------------------------------------------
// Kernel 4: causal flash attention. One wave (32 threads) per 16-query tile.
//   S = Q Kt (2x WMMA, K=32 chunks of D=64), online softmax with 16-lane
//   shuffle reductions, P restaged via LDS into an A-fragment, O += P V
//   (4x WMMA over D).  Grid = B*H*(T/16) = 8192 blocks.
// ---------------------------------------------------------------------------
__global__ void
__launch_bounds__(32)
attn_kernel(const _Float16* __restrict__ qbuf, const _Float16* __restrict__ kbuf,
            const _Float16* __restrict__ vtbuf, float* __restrict__ out) {
    __shared__ _Float16 ldsP[16 * 32];

    const int lane   = threadIdx.x;
    const int lane16 = lane & 15;
    const int offA   = (lane < 16) ? 0 : 16;
    const int offB   = (lane < 16) ? 0 : 16;
    const int rhi    = (lane < 16) ? 0 : 8;

    const int qt = blockIdx.x & 127;           // query tile (T/16 = 128)
    const int h  = (blockIdx.x >> 7) & 15;
    const int b  = blockIdx.x >> 11;
    const int q0 = qt * 16;
    const size_t bh = (size_t)(b * HH + h);

    const _Float16* qp = qbuf  + (bh * TT) * DD;
    const _Float16* kp = kbuf  + (bh * TT) * DD;
    const _Float16* vp = vtbuf + (bh * DD) * TT;

    // Q A-fragments for the two K=32 chunks of D (loaded once)
    v16h aq[2];
#pragma unroll
    for (int c = 0; c < 2; ++c) {
        const _Float16* qrow = qp + (size_t)(q0 + lane16) * DD + c * 32;
        aq[c] = cat16(*(const v8h*)(qrow + offA), *(const v8h*)(qrow + 8 + offA));
    }

    v8f O[4];
#pragma unroll
    for (int j = 0; j < 4; ++j)
#pragma unroll
        for (int e = 0; e < 8; ++e) O[j][e] = 0.0f;

    float mrow[8], lrow[8];
#pragma unroll
    for (int e = 0; e < 8; ++e) { mrow[e] = -1e30f; lrow[e] = 0.0f; }

    const float scale = 0.125f; // 1/sqrt(64)
    const int nkt = ((q0 + 15) >> 5) + 1; // causal: 32-key tiles needed

    for (int kt = 0; kt < nkt; ++kt) {
        const int k0 = kt * 32;

        // ---- load all 4 K-fragments up front, then S = Q Kt ----
        v16h bk[2][2];
#pragma unroll
        for (int c = 0; c < 2; ++c) {
            bk[c][0] = *(const v16h*)(kp + (size_t)(k0 + lane16) * DD + c * 32 + offB);
            bk[c][1] = *(const v16h*)(kp + (size_t)(k0 + 16 + lane16) * DD + c * 32 + offB);
        }
        v8f s0, s1;
#pragma unroll
        for (int e = 0; e < 8; ++e) { s0[e] = 0.0f; s1[e] = 0.0f; }
#pragma unroll
        for (int c = 0; c < 2; ++c) {
            s0 = __builtin_amdgcn_wmma_f32_16x16x32_f16(false, aq[c], false, bk[c][0],
                                                        (short)0, s0, false, false);
            s1 = __builtin_amdgcn_wmma_f32_16x16x32_f16(false, aq[c], false, bk[c][1],
                                                        (short)0, s1, false, false);
        }

        // ---- online softmax (per row e; row values live across 16 lanes) ----
#pragma unroll
        for (int e = 0; e < 8; ++e) {
            int row  = q0 + e + rhi;
            int col0 = k0 + lane16;
            int col1 = col0 + 16;
            float v0 = s0[e] * scale; if (col0 > row) v0 = -1e30f;
            float v1 = s1[e] * scale; if (col1 > row) v1 = -1e30f;

            float rmax = fmaxf(v0, v1);
#pragma unroll
            for (int off = 1; off < 16; off <<= 1)
                rmax = fmaxf(rmax, __shfl_xor(rmax, off, 32));

            float mnew = fmaxf(mrow[e], rmax);
            float corr = __expf(mrow[e] - mnew);
            float p0 = __expf(v0 - mnew);
            float p1 = __expf(v1 - mnew);
            float rsum = p0 + p1;
#pragma unroll
            for (int off = 1; off < 16; off <<= 1)
                rsum += __shfl_xor(rsum, off, 32);

            lrow[e] = lrow[e] * corr + rsum;
            mrow[e] = mnew;
#pragma unroll
            for (int j = 0; j < 4; ++j) O[j][e] *= corr;

            int rl = e + rhi;
            ldsP[rl * 32 + lane16]      = (_Float16)p0;
            ldsP[rl * 32 + 16 + lane16] = (_Float16)p1;
        }
        __syncthreads();

        // ---- P as A-fragment (rows = queries, K = 32 keys) ----
        v16h pf = cat16(*(const v8h*)&ldsP[lane16 * 32 + offA],
                        *(const v8h*)&ldsP[lane16 * 32 + 8 + offA]);
        __syncthreads();

        // ---- O += P V  (V^T gives K-contiguous B fragments) ----
#pragma unroll
        for (int j = 0; j < 4; ++j) {
            v16h bv = *(const v16h*)(vp + (size_t)(j * 16 + lane16) * TT + k0 + offB);
            O[j] = __builtin_amdgcn_wmma_f32_16x16x32_f16(false, pf, false, bv,
                                                          (short)0, O[j], false, false);
        }
    }

    // ---- normalize and write out [B,T,C] with head interleave ----
#pragma unroll
    for (int e = 0; e < 8; ++e) {
        float inv = 1.0f / lrow[e];
        int t = q0 + e + rhi;
        size_t base = ((size_t)b * TT + t) * CC + h * DD;
#pragma unroll
        for (int j = 0; j < 4; ++j)
            out[base + j * 16 + lane16] = O[j][e] * inv;
    }
}

// ---------------------------------------------------------------------------
extern "C" void kernel_launch(void* const* d_in, const int* in_sizes, int n_in,
                              void* d_out, int out_size, void* d_ws, size_t ws_size,
                              hipStream_t stream) {
    const float* x    = (const float*)d_in[0]; // [B,T,C]
    const float* wq   = (const float*)d_in[1]; // [C,3C]
    const float* bq   = (const float*)d_in[2]; // [3C]
    float*       outp = (float*)d_out;         // [B,T,C]

    // Workspace carve-up (f16 staging, ~74 MB total)
    char* ws = (char*)d_ws;
    _Float16* xh   = (_Float16*)(ws);
    _Float16* wt   = (_Float16*)(ws + (size_t)16777216);
    _Float16* kbuf = (_Float16*)(ws + (size_t)16777216 + 6291456);
    _Float16* qbuf = (_Float16*)(ws + (size_t)16777216 + 6291456 + 16777216);
    _Float16* vt   = (_Float16*)(ws + (size_t)16777216 + 6291456 + 2 * 16777216);

    // 1) x -> f16   (8,388,608 elems, 4/thread)
    cvt_x_kernel<<<8192, 256, 0, stream>>>(x, xh);
    // 2) W -> Wt f16 (3,145,728 elems)
    cvt_wt_kernel<<<12288, 256, 0, stream>>>(wq, wt);
    // 3) QKV GEMM + split/transpose routing
    qkv_gemm_kernel<<<dim3(64, 48), 128, 0, stream>>>(xh, wt, bq, kbuf, qbuf, vt);
    // 4) causal flash attention
    attn_kernel<<<BB * HH * (TT / 16), 32, 0, stream>>>(qbuf, kbuf, vt, outp);
}
